// MetadataAgePredictor_20598663151935
// MI455X (gfx1250) — compile-verified, matrix operand order checked by reading, not run on hardware
//
#include <hip/hip_runtime.h>
#include <hip/hip_bf16.h>
#include <cmath>

typedef __attribute__((ext_vector_type(16))) _Float16 v16h;
typedef __attribute__((ext_vector_type(8)))  float    v8f;

#define WAVES      8
#define AB_STRIDE  66            // 16x66 f32 activation buffer (bank-spread)
#define CB_STRIDE  36            // 16x36 f32 context buffer (latent32 + prot)
#define NFRAGS     80
#define FRAG_HALFS 512
#define WLDS_BYTES (NFRAGS * FRAG_HALFS * 2)   // 81920 B of f16 weight fragments

__device__ __forceinline__ v8f splat8(float x) { v8f c;
#pragma unroll
  for (int i = 0; i < 8; ++i) c[i] = x; return c; }

__device__ __forceinline__ v8f wmma_f16(v16h a, v16h b, v8f c) {
  return __builtin_amdgcn_wmma_f32_16x16x32_f16(false, a, false, b, (short)0, c, false, false);
}

// Inline erf (Abramowitz-Stegun 7.1.26, |err| < 1.5e-7) using HW v_exp_f32.
__device__ __forceinline__ float fast_erff(float x) {
  float a = fabsf(x);
  float t = 1.0f / fmaf(0.3275911f, a, 1.0f);
  float p = fmaf(t, 1.061405429f, -1.453152027f);
  p = fmaf(t, p, 1.421413741f);
  p = fmaf(t, p, -0.284496736f);
  p = fmaf(t, p, 0.254829592f);
  float r = 1.0f - t * p * __expf(-a * a);
  return copysignf(r, x);
}

__device__ __forceinline__ float gelu_exact(float v) {
  return 0.5f * v * (1.0f + fast_erff(v * 0.70710678118654752f));
}

__device__ __forceinline__ float softplusf(float x) {
  return (x > 20.0f) ? x : __logf(1.0f + __expf(x));
}

// Stage a K x N f32 weight into f16 WMMA B-fragments (ktiles x ntiles frags of 512 halves).
// Element (lane,i) of a frag: n = nt*16 + (lane&15), k = kt*32 + ((i&8)<<1) + ((lane&16)>>1) + (i&7)
__device__ __forceinline__ void stage_frag_group(_Float16* dst, const float* W, int ld, int Krows,
                                                 int Ncols, int ktiles, int ntiles, int tid, int nthr) {
  int total = ktiles * ntiles * FRAG_HALFS;
  for (int idx = tid; idx < total; idx += nthr) {
    int frag = idx >> 9;
    int kt = frag / ntiles;
    int nt = frag - kt * ntiles;
    int w = idx & 511;
    int lane = w >> 4;
    int i = w & 15;
    int n = nt * 16 + (lane & 15);
    int k = kt * 32 + ((i & 8) << 1) + ((lane & 16) >> 1) + (i & 7);
    float v = (k < Krows && n < Ncols) ? W[(long)k * ld + n] : 0.0f;
    dst[idx] = (_Float16)v;
  }
}

// Load A fragment (16 rows x 32 K, f32 LDS -> f16 regs) in WMMA A layout.
// Two-phase (load all, then convert) so the LDS loads can be issued back-to-back.
__device__ __forceinline__ v16h load_A_frag(const float* buf, int stride, int kbase, int lane) {
  int r = lane & 15;
  int kb = kbase + ((lane & 16) >> 1);
  const float* p = buf + r * stride;
  float tmp[16];
#pragma unroll
  for (int i = 0; i < 16; ++i) {
    int k = kb + ((i & 8) << 1) + (i & 7);
    tmp[i] = p[k];
  }
  v16h a;
#pragma unroll
  for (int i = 0; i < 16; ++i) a[i] = (_Float16)tmp[i];
  return a;
}

__device__ __forceinline__ v16h load_B_frag(const _Float16* wfrag, int fragIdx, int lane) {
  const _Float16* p = wfrag + fragIdx * FRAG_HALFS + lane * 16;
  v16h b;
#pragma unroll
  for (int i = 0; i < 16; ++i) b[i] = p[i];
  return b;
}

// LayerNorm(64) + exact GELU on a wave-private 16x64 f32 tile.
__device__ __forceinline__ void ln_gelu(float* buf, const float* g, const float* b, int lane) {
  int r = lane & 15;
  int c0 = (lane & 16) ? 32 : 0;
  float* row = buf + r * AB_STRIDE;
  float s = 0.0f, s2 = 0.0f;
#pragma unroll
  for (int c = 0; c < 32; ++c) { float v = row[c0 + c]; s += v; s2 += v * v; }
  s  += __shfl_xor(s, 16, 32);
  s2 += __shfl_xor(s2, 16, 32);
  float mu  = s * 0.015625f;
  float var = s2 * 0.015625f - mu * mu;
  float rstd = rsqrtf(var + 1e-5f);
#pragma unroll
  for (int c = 0; c < 32; ++c) {
    int col = c0 + c;
    float v = (row[col] - mu) * rstd * g[col] + b[col];
    row[col] = gelu_exact(v);
  }
}

__global__ __launch_bounds__(256)
void fused_flow_kernel(const float* __restrict__ metadata, const float* __restrict__ prot,
                       const float* __restrict__ age, const float* __restrict__ maskp,
                       const float* __restrict__ eW1, const float* __restrict__ eb1,
                       const float* __restrict__ lg1, const float* __restrict__ lb1,
                       const float* __restrict__ eW2, const float* __restrict__ eb2,
                       const float* __restrict__ lg2, const float* __restrict__ lb2,
                       const float* __restrict__ eW3, const float* __restrict__ eb3,
                       const float* __restrict__ fW1, const float* __restrict__ fb1,
                       const float* __restrict__ fW2, const float* __restrict__ fb2,
                       const float* __restrict__ fW3, const float* __restrict__ fb3,
                       float* __restrict__ out_latent, float* __restrict__ out_lp,
                       long B) {
  extern __shared__ char smem[];
  _Float16* wfrag = (_Float16*)smem;
  int tid = threadIdx.x;

  // ---- stage all weights into WMMA fragment layout (once per block) ----
  stage_frag_group(wfrag + 0  * FRAG_HALFS, eW1, 64, 22, 64, 1, 4, tid, 256);
  stage_frag_group(wfrag + 4  * FRAG_HALFS, eW2, 64, 64, 64, 2, 4, tid, 256);
  stage_frag_group(wfrag + 12 * FRAG_HALFS, eW3, 32, 64, 32, 2, 2, tid, 256);
#pragma unroll
  for (int t = 0; t < 4; ++t) {
    int fb = 16 + t * 16;
    stage_frag_group(wfrag + (fb + 0)  * FRAG_HALFS, fW1 + (long)t * 33 * 64, 64, 32, 64, 1, 4, tid, 256);
    stage_frag_group(wfrag + (fb + 4)  * FRAG_HALFS, fW2 + (long)t * 64 * 64, 64, 64, 64, 2, 4, tid, 256);
    stage_frag_group(wfrag + (fb + 12) * FRAG_HALFS, fW3 + (long)t * 64 * 23, 23, 64, 23, 2, 2, tid, 256);
  }
  __syncthreads();

  int wave = tid >> 5, lane = tid & 31;
  float* abuf = (float*)(smem + WLDS_BYTES) + wave * (16 * AB_STRIDE + 16 * CB_STRIDE);
  float* cbuf = abuf + 16 * AB_STRIDE;

  long tile = (long)blockIdx.x * WAVES + wave;
  long row0 = tile * 16;
  if (row0 >= B) return;

  int r_  = lane & 15;
  int mb  = (lane & 16) ? 8 : 0;
  int nlo = lane & 15;

  // ---- stage input tile: [metadata | mask | zero-pad] -> 16x32 ----
  for (int idx = lane; idx < 176; idx += 32) {
    int r = idx / 11, c = idx - r * 11;
    long row = row0 + r; if (row >= B) row = B - 1;
    abuf[r * AB_STRIDE + c]      = metadata[row * 11 + c];
    abuf[r * AB_STRIDE + 11 + c] = maskp[row * 11 + c];
  }
  for (int idx = lane; idx < 160; idx += 32) {
    int r = idx / 10, c = idx - r * 10;
    abuf[r * AB_STRIDE + 22 + c] = 0.0f;
  }

  // ---- encoder layer 1: (16x32) @ (32x64), bias pre-loaded into accumulator ----
  v16h a0 = load_A_frag(abuf, AB_STRIDE, 0, lane);
  v16h a1;
  {
    v8f acc[4];
#pragma unroll
    for (int nt = 0; nt < 4; ++nt) {
      v8f c = splat8(eb1[nt * 16 + nlo]);
      acc[nt] = wmma_f16(a0, load_B_frag(wfrag, nt, lane), c);
    }
#pragma unroll
    for (int nt = 0; nt < 4; ++nt)
#pragma unroll
      for (int j = 0; j < 8; ++j) abuf[(mb + j) * AB_STRIDE + nt * 16 + nlo] = acc[nt][j];
  }
  ln_gelu(abuf, lg1, lb1, lane);

  // ---- encoder layer 2: (16x64) @ (64x64) ----
  a0 = load_A_frag(abuf, AB_STRIDE, 0, lane);
  a1 = load_A_frag(abuf, AB_STRIDE, 32, lane);
  {
    v8f acc[4];
#pragma unroll
    for (int nt = 0; nt < 4; ++nt) {
      v8f c = splat8(eb2[nt * 16 + nlo]);
      c = wmma_f16(a0, load_B_frag(wfrag, 4 + nt, lane), c);
      acc[nt] = wmma_f16(a1, load_B_frag(wfrag, 8 + nt, lane), c);
    }
#pragma unroll
    for (int nt = 0; nt < 4; ++nt)
#pragma unroll
      for (int j = 0; j < 8; ++j) abuf[(mb + j) * AB_STRIDE + nt * 16 + nlo] = acc[nt][j];
  }
  ln_gelu(abuf, lg2, lb2, lane);

  // ---- encoder layer 3 -> latent (16x32), write out + build context ----
  a0 = load_A_frag(abuf, AB_STRIDE, 0, lane);
  a1 = load_A_frag(abuf, AB_STRIDE, 32, lane);
#pragma unroll
  for (int nt = 0; nt < 2; ++nt) {
    v8f c = splat8(eb3[nt * 16 + nlo]);
    c = wmma_f16(a0, load_B_frag(wfrag, 12 + nt, lane), c);
    c = wmma_f16(a1, load_B_frag(wfrag, 14 + nt, lane), c);
#pragma unroll
    for (int j = 0; j < 8; ++j) {
      int m = mb + j;
      cbuf[m * CB_STRIDE + nt * 16 + nlo] = c[j];
      long row = row0 + m;
      if (row < B) out_latent[row * 32 + nt * 16 + nlo] = c[j];
    }
  }
  for (int idx = lane; idx < 16; idx += 32) {
    long row = row0 + idx; if (row >= B) row = B - 1;
    cbuf[idx * CB_STRIDE + 32] = prot[row];
  }

  // hoist context A-fragment + per-row prot values (rank-1 term for W1 row 32)
  v16h ca = load_A_frag(cbuf, CB_STRIDE, 0, lane);
  float pr[8];
#pragma unroll
  for (int j = 0; j < 8; ++j) pr[j] = cbuf[(mb + j) * CB_STRIDE + 32];

  float z = 0.0f, ladj = 0.0f;
  {
    long row = row0 + r_; if (row >= B) row = B - 1;
    z = age[row];
  }

  // ---- 4 flow transforms ----
  for (int t = 0; t < 4; ++t) {
    int fbase = 16 + t * 16;
    // p1 = relu(context @ W1 + b1); rank-1 prot term folded into accumulator init
    {
      v8f acc[4];
#pragma unroll
      for (int nt = 0; nt < 4; ++nt) {
        float w32  = fW1[((long)t * 33 + 32) * 64 + nt * 16 + nlo];
        float bias = fb1[t * 64 + nt * 16 + nlo];
        v8f c;
#pragma unroll
        for (int j = 0; j < 8; ++j) c[j] = fmaf(pr[j], w32, bias);
        acc[nt] = wmma_f16(ca, load_B_frag(wfrag, fbase + nt, lane), c);
      }
#pragma unroll
      for (int nt = 0; nt < 4; ++nt)
#pragma unroll
        for (int j = 0; j < 8; ++j)
          abuf[(mb + j) * AB_STRIDE + nt * 16 + nlo] = fmaxf(acc[nt][j], 0.0f);
    }
    // p2 = relu(p1 @ W2 + b2)
    a0 = load_A_frag(abuf, AB_STRIDE, 0, lane);
    a1 = load_A_frag(abuf, AB_STRIDE, 32, lane);
    {
      v8f acc[4];
#pragma unroll
      for (int nt = 0; nt < 4; ++nt) {
        v8f c = splat8(fb2[t * 64 + nt * 16 + nlo]);
        c = wmma_f16(a0, load_B_frag(wfrag, fbase + 4 + nt, lane), c);
        acc[nt] = wmma_f16(a1, load_B_frag(wfrag, fbase + 8 + nt, lane), c);
      }
#pragma unroll
      for (int nt = 0; nt < 4; ++nt)
#pragma unroll
        for (int j = 0; j < 8; ++j)
          abuf[(mb + j) * AB_STRIDE + nt * 16 + nlo] = fmaxf(acc[nt][j], 0.0f);
    }
    // p3 = p2 @ W3 + b3  (23 cols)
    a0 = load_A_frag(abuf, AB_STRIDE, 0, lane);
    a1 = load_A_frag(abuf, AB_STRIDE, 32, lane);
#pragma unroll
    for (int nt = 0; nt < 2; ++nt) {
      int col = nt * 16 + nlo;
      v8f c = splat8((col < 23) ? fb3[t * 23 + col] : 0.0f);
      c = wmma_f16(a0, load_B_frag(wfrag, fbase + 12 + nt, lane), c);
      c = wmma_f16(a1, load_B_frag(wfrag, fbase + 14 + nt, lane), c);
#pragma unroll
      for (int j = 0; j < 8; ++j) abuf[(mb + j) * AB_STRIDE + col] = c[j];
    }
    // ---- rational-quadratic spline, one row per lane (lanes 0..15) ----
    if (lane < 16) {
      const float* prow = abuf + r_ * AB_STRIDE;
      float wv[8], hv[8];
      float mw = -1e30f, mh = -1e30f;
#pragma unroll
      for (int k = 0; k < 8; ++k) {
        wv[k] = prow[k]; hv[k] = prow[8 + k];
        mw = fmaxf(mw, wv[k]); mh = fmaxf(mh, hv[k]);
      }
      float sw = 0.0f, sh = 0.0f;
#pragma unroll
      for (int k = 0; k < 8; ++k) {
        wv[k] = __expf(wv[k] - mw); sw += wv[k];
        hv[k] = __expf(hv[k] - mh); sh += hv[k];
      }
      float iw = 10.0f / sw, ih = 10.0f / sh;   // 2*bound / softmax sum
#pragma unroll
      for (int k = 0; k < 8; ++k) { wv[k] *= iw; hv[k] *= ih; }

      bool inside = (z > -5.0f) && (z < 5.0f);
      float xc = fminf(fmaxf(z, -5.0f), 5.0f);
      float xe = -5.0f, ye = -5.0f;
      float X0 = -5.0f, Y0 = -5.0f, wk = wv[0], hk = hv[0], d0 = 1.0f, d1 = 1.0f, dj = 1.0f;
#pragma unroll
      for (int k = 0; k < 8; ++k) {
        float dnext = (k < 7) ? softplusf(prow[16 + k]) : 1.0f;
        if (xc >= xe) { X0 = xe; Y0 = ye; wk = wv[k]; hk = hv[k]; d0 = dj; d1 = dnext; }
        xe += wv[k]; ye += hv[k];
        dj = dnext;
      }
      float s  = hk / wk;
      float xi = (xc - X0) / wk;
      float om = 1.0f - xi;
      float den = s + (d0 + d1 - 2.0f * s) * xi * om;
      float y_in = Y0 + hk * (s * xi * xi + d0 * xi * om) / den;
      float ld_in = 2.0f * __logf(s) + __logf(d1 * xi * xi + 2.0f * s * xi * om + d0 * om * om)
                    - 2.0f * __logf(den);
      if (inside) { z = y_in; ladj += ld_in; }
    }
  }

  if (lane < 16) {
    long row = row0 + r_;
    if (row < B) out_lp[row] = -0.5f * z * z - 0.91893853320467274f + ladj;
  }
}

__global__ __launch_bounds__(256)
void lp_partial_kernel(const float* __restrict__ lp, long n, float* __restrict__ partial) {
  __shared__ float sm[256];
  float s = 0.0f;
  for (long i = (long)blockIdx.x * 256 + threadIdx.x; i < n; i += (long)gridDim.x * 256) s += lp[i];
  sm[threadIdx.x] = s;
  __syncthreads();
  for (int off = 128; off > 0; off >>= 1) {
    if ((int)threadIdx.x < off) sm[threadIdx.x] += sm[threadIdx.x + off];
    __syncthreads();
  }
  if (threadIdx.x == 0) partial[blockIdx.x] = sm[0];
}

__global__ __launch_bounds__(256)
void lp_final_kernel(const float* __restrict__ partial, int np, float* __restrict__ out_nll, float invB) {
  __shared__ float sm[256];
  float s = 0.0f;
  for (int i = threadIdx.x; i < np; i += 256) s += partial[i];
  sm[threadIdx.x] = s;
  __syncthreads();
  for (int off = 128; off > 0; off >>= 1) {
    if ((int)threadIdx.x < off) sm[threadIdx.x] += sm[threadIdx.x + off];
    __syncthreads();
  }
  if (threadIdx.x == 0) out_nll[0] = -sm[0] * invB;
}

extern "C" void kernel_launch(void* const* d_in, const int* in_sizes, int n_in,
                              void* d_out, int out_size, void* d_ws, size_t ws_size,
                              hipStream_t stream) {
  const float* metadata = (const float*)d_in[0];
  const float* prot     = (const float*)d_in[1];
  const float* age      = (const float*)d_in[2];
  const float* maskp    = (const float*)d_in[3];
  const float* eW1 = (const float*)d_in[4];  const float* eb1 = (const float*)d_in[5];
  const float* lg1 = (const float*)d_in[6];  const float* lb1 = (const float*)d_in[7];
  const float* eW2 = (const float*)d_in[8];  const float* eb2 = (const float*)d_in[9];
  const float* lg2 = (const float*)d_in[10]; const float* lb2 = (const float*)d_in[11];
  const float* eW3 = (const float*)d_in[12]; const float* eb3 = (const float*)d_in[13];
  const float* fW1 = (const float*)d_in[14]; const float* fb1 = (const float*)d_in[15];
  const float* fW2 = (const float*)d_in[16]; const float* fb2 = (const float*)d_in[17];
  const float* fW3 = (const float*)d_in[18]; const float* fb3 = (const float*)d_in[19];

  long B = (long)in_sizes[1];
  float* out_latent = (float*)d_out;
  float* out_lp  = out_latent + B * 32;
  float* out_nll = out_lp + B;

  size_t lds = (size_t)WLDS_BYTES + (size_t)WAVES * (16 * AB_STRIDE + 16 * CB_STRIDE) * sizeof(float);
  (void)hipFuncSetAttribute((const void*)fused_flow_kernel,
                            hipFuncAttributeMaxDynamicSharedMemorySize, (int)lds);

  long tiles = (B + 15) / 16;
  int blocks = (int)((tiles + WAVES - 1) / WAVES);
  fused_flow_kernel<<<blocks, 256, lds, stream>>>(
      metadata, prot, age, maskp,
      eW1, eb1, lg1, lb1, eW2, eb2, lg2, lb2, eW3, eb3,
      fW1, fb1, fW2, fb2, fW3, fb3,
      out_latent, out_lp, B);

  int np = 1024;
  if (ws_size < (size_t)np * sizeof(float)) np = (int)(ws_size / sizeof(float));
  if (np < 1) np = 1;
  lp_partial_kernel<<<np, 256, 0, stream>>>(out_lp, B, (float*)d_ws);
  lp_final_kernel<<<1, 256, 0, stream>>>((const float*)d_ws, np, out_nll, 1.0f / (float)B);
}
